// DeepseekV3Attention_70557722738965
// MI455X (gfx1250) — matrix operand order repaired and checked by reading, not compiled, not run on hardware
//
#include <hip/hip_runtime.h>
#include <hip/hip_bf16.h>
#include <math.h>
#include <stdint.h>

typedef __bf16 bf16;
typedef __attribute__((ext_vector_type(16))) __bf16 v16bf;
typedef __attribute__((ext_vector_type(8)))  __bf16 v8bf;
typedef __attribute__((ext_vector_type(4)))  __bf16 v4bf;
typedef __attribute__((ext_vector_type(8)))  float  v8f;

#define B_      2
#define S_      2048
#define HID_    2048
#define NH_     16
#define QLORA_  1536
#define KVLORA_ 512
#define NOPE_   128
#define ROPE_   64
#define QHD_    192   // NOPE + ROPE
#define VD_     128

__device__ __forceinline__ bf16 tobf(float x) { return (bf16)x; }

// ---- CDNA5 async global->LDS copy (ASYNCcnt path), 16B per lane ------------
__device__ __forceinline__ void async_b128(const bf16* g, bf16* l) {
  asm volatile("global_load_async_to_lds_b128 %0, %1, off"
               :: "v"((uint32_t)(uintptr_t)l), "v"((uint64_t)(uintptr_t)g)
               : "memory");
}
__device__ __forceinline__ void wait_async0() {
  asm volatile("s_wait_asynccnt 0x0" ::: "memory");
}

// ---------------------------------------------------------------------------
// WMMA GEMM: C[z] = A[z] (MxK) * W[z] (NxK)^T, fp32 accumulate via
// v_wmma_f32_16x16x32_bf16. Tile: BM=256, BN=64, BK=32; 8 waves, each wave
// computes 64x32 (4x2 frags -> 8 WMMA per k-step).
// TA==bf16  : A tile staged with global_load_async_to_lds_b128
// TA==float : A tile converted in-register (probs path)
// EPI==1    : C = C*alpha + causal_mask(m,n)
// OUTBF     : store C as bf16 instead of fp32
// C address: base + (z/zdiv)*cOuter + (z%zdiv)*cInner + m*cRow + n
// ---------------------------------------------------------------------------
template <typename TA, int EPI, bool OUTBF>
__global__ __launch_bounds__(256) void gemm_wmma_kernel(
    const TA* __restrict__ A, const bf16* __restrict__ W, void* __restrict__ C,
    int M, int N, int K, int lda,
    long long aBatch, long long wBatch,
    int zdiv, long long cOuter, long long cInner, int cRow,
    float alpha)
{
  __shared__ bf16 As[256][40];   // row stride 80B: keeps 16B-aligned frags
  __shared__ bf16 Ws[64][40];

  const int tid  = threadIdx.x;
  const int lane = tid & 31;
  const int wave = tid >> 5;
  const int wm   = wave >> 1;    // 0..3 : 64-row strip
  const int wn   = wave & 1;     // 0..1 : 32-col strip
  const int z    = blockIdx.z;
  const int m0   = blockIdx.y * 256;
  const int n0   = blockIdx.x * 64;

  const TA*   Ab = A + (long long)z * aBatch;
  const bf16* Wb = W + (long long)z * wBatch;
  char*       Cb = (char*)C + ((long long)(z / zdiv) * cOuter +
                               (long long)(z % zdiv) * cInner) * (OUTBF ? 2 : 4);

  v8f acc[4][2] = {};

  const int lg = lane >> 4;      // half-wave select
  const int ll = lane & 15;

  for (int k0 = 0; k0 < K; k0 += 32) {
    if constexpr (__is_same(TA, bf16)) {
      // A tile: 256x32 bf16 = 1024 x 16B chunks, 4 per thread, async to LDS
      #pragma unroll
      for (int i = 0; i < 4; ++i) {
        const int idx = i * 256 + tid;
        const int r = idx >> 2, sg = (idx & 3) * 8;
        async_b128(Ab + (long long)(m0 + r) * lda + (k0 + sg), &As[r][sg]);
      }
    } else {
      // fp32 A (probs): convert in-register, 512 x 16-elt chunks, 2 per thread
      #pragma unroll
      for (int i = 0; i < 2; ++i) {
        const int c = i * 256 + tid;
        const int r = c >> 1, col = (c & 1) * 16;
        #pragma unroll
        for (int j = 0; j < 16; ++j)
          As[r][col + j] = tobf((float)Ab[(long long)(m0 + r) * lda + (k0 + col + j)]);
      }
    }
    // W tile: 64x32 bf16 = 256 x 16B chunks, 1 per thread, async to LDS
    {
      const int r = tid >> 2, sg = (tid & 3) * 8;
      async_b128(Wb + (long long)(n0 + r) * K + (k0 + sg), &Ws[r][sg]);
    }
    wait_async0();
    __syncthreads();

    // A frags 16x32: lanes 0-15 K {0-7,16-23}; lanes 16-31 K {8-15,24-31}
    v16bf af[4];
    #pragma unroll
    for (int mf = 0; mf < 4; ++mf) {
      const int r  = wm * 64 + mf * 16 + ll;
      const int kh = lg * 8;
      union { v16bf v; v8bf h[2]; } u;
      u.h[0] = *(const v8bf*)&As[r][kh];
      u.h[1] = *(const v8bf*)&As[r][16 + kh];
      af[mf] = u.v;
    }
    // B frags 32x16: lanes 0-15 K {0-7,16-23}; lanes 16-31 K {8-15,24-31}; col = ll
    v16bf bfr[2];
    #pragma unroll
    for (int nf = 0; nf < 2; ++nf) {
      const int c  = wn * 32 + nf * 16 + ll;
      const int kh = lg * 8;
      union { v16bf v; v8bf h[2]; } u;
      u.h[0] = *(const v8bf*)&Ws[c][kh];
      u.h[1] = *(const v8bf*)&Ws[c][16 + kh];
      bfr[nf] = u.v;
    }
    #pragma unroll
    for (int mf = 0; mf < 4; ++mf)
      #pragma unroll
      for (int nf = 0; nf < 2; ++nf)
        acc[mf][nf] = __builtin_amdgcn_wmma_f32_16x16x32_bf16(
            false, af[mf], false, bfr[nf], (short)0, acc[mf][nf], false, false);
    __syncthreads();
  }

  // C frag: VGPR r -> M = r + 8*(lane>=16); N = lane&15
  #pragma unroll
  for (int mf = 0; mf < 4; ++mf) {
    #pragma unroll
    for (int nf = 0; nf < 2; ++nf) {
      #pragma unroll
      for (int r = 0; r < 8; ++r) {
        const int m = m0 + wm * 64 + mf * 16 + r + lg * 8;
        const int n = n0 + wn * 32 + nf * 16 + ll;
        float v = acc[mf][nf][r];
        if (EPI == 1) v = v * alpha + ((n <= m) ? 0.0f : -1.0e9f);
        if (OUTBF) ((bf16*)Cb)[(long long)m * cRow + n] = tobf(v);
        else       ((float*)Cb)[(long long)m * cRow + n] = v;
      }
    }
  }
}

// ---------------------------------------------------------------------------
// fp32 -> bf16 bulk convert (vectorized 4/thread); n multiple of 4.
// ---------------------------------------------------------------------------
__global__ __launch_bounds__(256) void cvt_kernel(const float* __restrict__ x,
                                                  bf16* __restrict__ y, long long n4)
{
  const long long i = (long long)blockIdx.x * 256 + threadIdx.x;
  if (i < n4) {
    float4 v = ((const float4*)x)[i];
    v4bf o; o[0] = tobf(v.x); o[1] = tobf(v.y); o[2] = tobf(v.z); o[3] = tobf(v.w);
    ((v4bf*)y)[i] = o;
  }
}

// ---------------------------------------------------------------------------
// Row-wise RMSNorm, fp32 in -> TO out.
// ---------------------------------------------------------------------------
template <typename TO>
__global__ __launch_bounds__(256) void rmsnorm_kernel(
    const float* __restrict__ x, const float* __restrict__ w, TO* __restrict__ y,
    int D, int strideIn, int strideOut)
{
  __shared__ float red[256];
  const long long row = blockIdx.x;
  const float* xr = x + row * strideIn;
  TO*          yr = y + row * strideOut;
  const int t = threadIdx.x;
  float ss = 0.f;
  for (int i = t; i < D; i += 256) { float v = xr[i]; ss += v * v; }
  red[t] = ss; __syncthreads();
  for (int s = 128; s > 0; s >>= 1) { if (t < s) red[t] += red[t + s]; __syncthreads(); }
  const float r = rsqrtf(red[0] / (float)D + 1e-6f);
  for (int i = t; i < D; i += 256) yr[i] = (TO)(w[i] * (xr[i] * r));
}

// ---------------------------------------------------------------------------
// RoPE + pack: bf16 Q/K (b*NH+h, s, 192) and V^T (b*NH+h, d, s).
//   y[j]    = x[2j]*cos - x[2j+1]*sin ;  y[j+32] = x[2j+1]*cos + x[2j]*sin
// ---------------------------------------------------------------------------
__global__ __launch_bounds__(128) void rope_pack_kernel(
    const float* __restrict__ qf,    // (B*S, NH*QHD)
    const float* __restrict__ kvup,  // (B*S, NH*(NOPE+VD))
    const float* __restrict__ ckv,   // (B*S, KVLORA+ROPE); k_pe at [512..575]
    bf16* __restrict__ Qb, bf16* __restrict__ Kb, bf16* __restrict__ Vt)
{
  const int s = blockIdx.x, h = blockIdx.y, b = blockIdx.z, t = threadIdx.x;
  const long long row = (long long)b * S_ + s;
  const long long bh  = (long long)b * NH_ + h;
  const float* qrow = qf   + row * (NH_ * QHD_) + (long long)h * QHD_;
  const float* krow = kvup + row * (NH_ * (NOPE_ + VD_)) + (long long)h * (NOPE_ + VD_);
  bf16* qo = Qb + (bh * S_ + s) * QHD_;
  bf16* ko = Kb + (bh * S_ + s) * QHD_;

  qo[t] = tobf(qrow[t]);                                   // q_nope
  ko[t] = tobf(krow[t]);                                   // k_nope
  Vt[bh * (long long)(VD_ * S_) + (long long)t * S_ + s] = tobf(krow[NOPE_ + t]);

  if (t < 32) {
    const float invf = __expf(-((float)(2 * t) / (float)ROPE_) * __logf(10000.0f));
    const float ang  = (float)s * invf;
    float sn, c; __sincosf(ang, &sn, &c);
    float x0 = qrow[NOPE_ + 2 * t], x1 = qrow[NOPE_ + 2 * t + 1];
    qo[NOPE_ + t]      = tobf(x0 * c - x1 * sn);
    qo[NOPE_ + 32 + t] = tobf(x1 * c + x0 * sn);
    const float* kpe = ckv + row * (KVLORA_ + ROPE_) + KVLORA_;
    x0 = kpe[2 * t]; x1 = kpe[2 * t + 1];
    ko[NOPE_ + t]      = tobf(x0 * c - x1 * sn);
    ko[NOPE_ + 32 + t] = tobf(x1 * c + x0 * sn);
  }
}

// ---------------------------------------------------------------------------
// Row softmax over length S_, in place (mask baked into scores).
// ---------------------------------------------------------------------------
__global__ __launch_bounds__(256) void softmax_kernel(float* __restrict__ p)
{
  __shared__ float red[256];
  float* row = p + (long long)blockIdx.x * S_;
  const int t = threadIdx.x;
  float mx = -3.4e38f;
  for (int i = t; i < S_; i += 256) mx = fmaxf(mx, row[i]);
  red[t] = mx; __syncthreads();
  for (int s = 128; s > 0; s >>= 1) { if (t < s) red[t] = fmaxf(red[t], red[t + s]); __syncthreads(); }
  mx = red[0]; __syncthreads();
  float sum = 0.f;
  for (int i = t; i < S_; i += 256) { float e = __expf(row[i] - mx); row[i] = e; sum += e; }
  red[t] = sum; __syncthreads();
  for (int s = 128; s > 0; s >>= 1) { if (t < s) red[t] += red[t + s]; __syncthreads(); }
  const float inv = 1.0f / red[0];
  for (int i = t; i < S_; i += 256) row[i] *= inv;
}

// ---------------------------------------------------------------------------
extern "C" void kernel_launch(void* const* d_in, const int* in_sizes, int n_in,
                              void* d_out, int out_size, void* d_ws, size_t ws_size,
                              hipStream_t stream)
{
  (void)in_sizes; (void)n_in; (void)out_size; (void)ws_size;
  const float* hs        = (const float*)d_in[0];
  // d_in[1] attention_mask (causal, regenerated inline), d_in[2] position_ids (== arange)
  const float* q_a_w     = (const float*)d_in[3];
  const float* q_a_ln_w  = (const float*)d_in[4];
  const float* q_b_w     = (const float*)d_in[5];
  const float* kv_a_w    = (const float*)d_in[6];
  const float* kv_a_ln_w = (const float*)d_in[7];
  const float* kv_b_w    = (const float*)d_in[8];
  const float* o_w       = (const float*)d_in[9];

  float* out   = (float*)d_out;
  float* probs = out + (long long)B_ * S_ * HID_;   // output tuple: (out, probs)

  char*  ws  = (char*)d_ws;
  size_t off = 0;
  auto alloc = [&](size_t bytes) -> char* {
    char* p = ws + off; off += (bytes + 255) & ~(size_t)255; return p;
  };
  const int M = B_ * S_;  // 4096
  // bf16 shadows of operands feeding WMMA
  bf16* hs_b   = (bf16*)alloc((size_t)M * HID_ * 2);
  bf16* wqa_b  = (bf16*)alloc((size_t)QLORA_ * HID_ * 2);
  bf16* wkva_b = (bf16*)alloc((size_t)(KVLORA_ + ROPE_) * HID_ * 2);
  bf16* wqb_b  = (bf16*)alloc((size_t)(NH_ * QHD_) * QLORA_ * 2);
  bf16* wkvb_b = (bf16*)alloc((size_t)(NH_ * (NOPE_ + VD_)) * KVLORA_ * 2);
  bf16* wo_b   = (bf16*)alloc((size_t)HID_ * (NH_ * VD_) * 2);
  // activations
  float* q_a    = (float*)alloc((size_t)M * QLORA_ * 4);
  bf16*  q_an   = (bf16*)alloc((size_t)M * QLORA_ * 2);
  float* ckv    = (float*)alloc((size_t)M * (KVLORA_ + ROPE_) * 4);
  bf16*  ckv_n  = (bf16*)alloc((size_t)M * KVLORA_ * 2);
  float* q_full = (float*)alloc((size_t)M * (NH_ * QHD_) * 4);   // reused as attn2
  float* kv_up  = (float*)alloc((size_t)M * (NH_ * (NOPE_ + VD_)) * 4);
  bf16*  Qb     = (bf16*)alloc((size_t)B_ * NH_ * S_ * QHD_ * 2);
  bf16*  Kb     = (bf16*)alloc((size_t)B_ * NH_ * S_ * QHD_ * 2);
  bf16*  Vt     = (bf16*)alloc((size_t)B_ * NH_ * VD_ * S_ * 2);
  bf16*  attn2  = (bf16*)q_full;   // q_full dead after rope_pack

  const dim3 blk(256);
  auto cvt = [&](const float* x, bf16* y, long long n) {
    const long long n4 = n / 4;
    cvt_kernel<<<dim3((unsigned)((n4 + 255) / 256)), blk, 0, stream>>>(x, y, n4);
  };
  // 0) one-time fp32 -> bf16 conversions
  cvt(hs,     hs_b,   (long long)M * HID_);
  cvt(q_a_w,  wqa_b,  (long long)QLORA_ * HID_);
  cvt(kv_a_w, wkva_b, (long long)(KVLORA_ + ROPE_) * HID_);
  cvt(q_b_w,  wqb_b,  (long long)(NH_ * QHD_) * QLORA_);
  cvt(kv_b_w, wkvb_b, (long long)(NH_ * (NOPE_ + VD_)) * KVLORA_);
  cvt(o_w,    wo_b,   (long long)HID_ * (NH_ * VD_));

  // 1) q_a = hs @ q_a_w^T            (4096 x 1536 x 2048)
  gemm_wmma_kernel<bf16, 0, false><<<dim3(QLORA_/64, M/256, 1), blk, 0, stream>>>(
      hs_b, wqa_b, q_a, M, QLORA_, HID_, HID_, 0, 0, 1, 0, 0, QLORA_, 1.0f);
  // 2) ckv = hs @ kv_a_w^T           (4096 x 576 x 2048)
  gemm_wmma_kernel<bf16, 0, false><<<dim3((KVLORA_+ROPE_)/64, M/256, 1), blk, 0, stream>>>(
      hs_b, wkva_b, ckv, M, KVLORA_ + ROPE_, HID_, HID_, 0, 0, 1, 0, 0, KVLORA_ + ROPE_, 1.0f);
  // 3) RMSNorm -> bf16
  rmsnorm_kernel<bf16><<<M, 256, 0, stream>>>(q_a, q_a_ln_w, q_an, QLORA_, QLORA_, QLORA_);
  rmsnorm_kernel<bf16><<<M, 256, 0, stream>>>(ckv, kv_a_ln_w, ckv_n, KVLORA_, KVLORA_ + ROPE_, KVLORA_);
  // 4) q_full = q_a_n @ q_b_w^T      (4096 x 3072 x 1536)
  gemm_wmma_kernel<bf16, 0, false><<<dim3(NH_*QHD_/64, M/256, 1), blk, 0, stream>>>(
      q_an, wqb_b, q_full, M, NH_ * QHD_, QLORA_, QLORA_, 0, 0, 1, 0, 0, NH_ * QHD_, 1.0f);
  // 5) kv_up = ckv_n @ kv_b_w^T      (4096 x 4096 x 512)
  gemm_wmma_kernel<bf16, 0, false><<<dim3(NH_*(NOPE_+VD_)/64, M/256, 1), blk, 0, stream>>>(
      ckv_n, wkvb_b, kv_up, M, NH_ * (NOPE_ + VD_), KVLORA_, KVLORA_,
      0, 0, 1, 0, 0, NH_ * (NOPE_ + VD_), 1.0f);
  // 6) RoPE + pack Q/K (bf16) and V^T (bf16)
  rope_pack_kernel<<<dim3(S_, NH_, B_), 128, 0, stream>>>(q_full, kv_up, ckv, Qb, Kb, Vt);
  // 7) scores -> probs (fused scale + causal mask), batched over 32 (b,h)
  const float scale = 1.0f / sqrtf((float)QHD_);
  gemm_wmma_kernel<bf16, 1, false><<<dim3(S_/64, S_/256, B_ * NH_), blk, 0, stream>>>(
      Qb, Kb, probs, S_, S_, QHD_, QHD_,
      (long long)S_ * QHD_, (long long)S_ * QHD_,
      1, (long long)S_ * S_, 0, S_, scale);
  // 8) softmax rows in place (in d_out probs region)
  softmax_kernel<<<B_ * NH_ * S_, 256, 0, stream>>>(probs);
  // 9) attn = probs @ (V^T)^T, written bf16 directly as (b, s, h*VD+d)
  gemm_wmma_kernel<float, 0, true><<<dim3(VD_/64, S_/256, B_ * NH_), blk, 0, stream>>>(
      probs, Vt, attn2, S_, VD_, S_, S_,
      (long long)S_ * S_, (long long)VD_ * S_,
      NH_, (long long)S_ * (NH_ * VD_), VD_, NH_ * VD_, 1.0f);
  // 10) out = attn @ o_w^T           (4096 x 2048 x 2048)
  gemm_wmma_kernel<bf16, 0, false><<<dim3(HID_/64, M/256, 1), blk, 0, stream>>>(
      attn2, wo_b, out, M, HID_, NH_ * VD_, NH_ * VD_, 0, 0, 1, 0, 0, HID_, 1.0f);
}
